// AttnPooling_11819749998824
// MI455X (gfx1250) — matrix operand order repaired and checked by reading, not compiled, not run on hardware
//
#include <hip/hip_runtime.h>
#include <hip/hip_bf16.h>

typedef __attribute__((ext_vector_type(16))) __bf16 v16bf;
typedef __attribute__((ext_vector_type(8)))  __bf16 v8bf;
typedef __attribute__((ext_vector_type(8)))  float   v8f;

#define B_   64
#define N_   256
#define D_   1024
#define H_   16
#define DK_  64
#define EPS_ 1e-5f

__device__ inline v8f zero_v8f() {
  v8f z;
#pragma unroll
  for (int i = 0; i < 8; i++) z[i] = 0.f;
  return z;
}

__device__ inline v16bf cat8(v8bf lo, v8bf hi) {
  return __builtin_shufflevector(lo, hi, 0, 1, 2, 3, 4, 5, 6, 7,
                                 8, 9, 10, 11, 12, 13, 14, 15);
}

// ---------------------------------------------------------------------------
// Weight prep: Wt[n][k] (bf16) = W[k][n] (fp32). 64x64 LDS-tiled transpose.
// Grid (16,16) per 1024x1024 matrix.
// ---------------------------------------------------------------------------
__global__ __launch_bounds__(256)
void wt_transpose_kernel(const float* __restrict__ W, __bf16* __restrict__ Wt) {
  __shared__ float tile[64][65];
  const int tid = threadIdx.x;
  const int k0 = blockIdx.y * 64, n0 = blockIdx.x * 64;
  const int c = tid & 63, rp = tid >> 6;
#pragma unroll
  for (int p = 0; p < 16; p++) {
    const int k = rp + p * 4;
    tile[k][c] = W[(size_t)(k0 + k) * D_ + n0 + c];
  }
  __syncthreads();
#pragma unroll
  for (int p = 0; p < 16; p++) {
    const int n = rp + p * 4;
    Wt[(size_t)(n0 + n) * D_ + k0 + c] = (__bf16)tile[c][n];
  }
}

// ---------------------------------------------------------------------------
// Flat fp32 -> bf16 conversion (16M elements, 8 per thread).
// ---------------------------------------------------------------------------
__global__ __launch_bounds__(256)
void cvt_bf16_kernel(const float* __restrict__ X, __bf16* __restrict__ Y) {
  const size_t i = ((size_t)blockIdx.x * 256 + threadIdx.x) * 8;
  const float4 a = *(const float4*)(X + i);
  const float4 b = *(const float4*)(X + i + 4);
  v8bf o;
  o[0] = (__bf16)a.x; o[1] = (__bf16)a.y; o[2] = (__bf16)a.z; o[3] = (__bf16)a.w;
  o[4] = (__bf16)b.x; o[5] = (__bf16)b.y; o[6] = (__bf16)b.z; o[7] = (__bf16)b.w;
  *(v8bf*)(Y + i) = o;
}

// ---------------------------------------------------------------------------
// GEMM: acc[M,N] = A[M,K](bf16) * Wt[N,K](bf16)^T + bias.
//   outf (fp32, nullable): acc + bias (+ resid)
//   outb (bf16, nullable): same value as bf16
// bf16 WMMA 16x16x32, fp32 accumulate. 128x128x32 tiles, double-buffered LDS.
// Tile staging is pure b128 copy (both operands pre-converted / pre-transposed).
// ---------------------------------------------------------------------------
__global__ __launch_bounds__(256)
void gemm_bias_kernel(const __bf16* __restrict__ A, const __bf16* __restrict__ Wt,
                      const float* __restrict__ bias, const float* __restrict__ resid,
                      float* __restrict__ outf, __bf16* __restrict__ outb,
                      int M, int N, int K) {
  constexpr int BM = 128, BN = 128, BK = 32, ST = BK + 8;
  __shared__ __bf16 As[2][BM * ST];   // [row][k]
  __shared__ __bf16 Bs[2][BN * ST];   // [n][k]
  const int tid = threadIdx.x, lane = tid & 31, wid = tid >> 5;
  const int wm = wid & 3, wn = wid >> 2, lr = lane & 15, hl = lane >> 4;
  const int bm0 = blockIdx.y * BM, bn0 = blockIdx.x * BN;

  v8f acc[2][4];
#pragma unroll
  for (int i = 0; i < 2; i++)
#pragma unroll
    for (int j = 0; j < 4; j++) acc[i][j] = zero_v8f();

  const int c8 = (tid & 3) * 8, r0 = tid >> 2;  // r0 in 0..63

  auto load_tiles = [&](int kk, int buf) {
    // Issue all global loads first so waits can be batched, then store to LDS.
    const v8bf a0 = *(const v8bf*)(A + (size_t)(bm0 + r0) * K + kk + c8);
    const v8bf a1 = *(const v8bf*)(A + (size_t)(bm0 + r0 + 64) * K + kk + c8);
    const v8bf b0 = *(const v8bf*)(Wt + (size_t)(bn0 + r0) * K + kk + c8);
    const v8bf b1 = *(const v8bf*)(Wt + (size_t)(bn0 + r0 + 64) * K + kk + c8);
    *(v8bf*)&As[buf][r0 * ST + c8] = a0;
    *(v8bf*)&As[buf][(r0 + 64) * ST + c8] = a1;
    *(v8bf*)&Bs[buf][r0 * ST + c8] = b0;
    *(v8bf*)&Bs[buf][(r0 + 64) * ST + c8] = b1;
  };

  load_tiles(0, 0);
  __syncthreads();
  const int nk = K / BK;
  for (int kb = 0; kb < nk; kb++) {
    const int buf = kb & 1;
    if (kb + 1 < nk) load_tiles((kb + 1) * BK, buf ^ 1);
    v16bf af[2], bfv[4];
#pragma unroll
    for (int t = 0; t < 2; t++) {
      const int row = wm * 32 + t * 16 + lr;
      v8bf lo = *(const v8bf*)&As[buf][row * ST + hl * 8];
      v8bf hb = *(const v8bf*)&As[buf][row * ST + 16 + hl * 8];
      af[t] = cat8(lo, hb);
    }
#pragma unroll
    for (int t = 0; t < 4; t++) {
      const int col = wn * 64 + t * 16 + lr;
      v8bf lo = *(const v8bf*)&Bs[buf][col * ST + hl * 8];
      v8bf hb = *(const v8bf*)&Bs[buf][col * ST + 16 + hl * 8];
      bfv[t] = cat8(lo, hb);
    }
#pragma unroll
    for (int tm = 0; tm < 2; tm++)
#pragma unroll
      for (int tn = 0; tn < 4; tn++)
        acc[tm][tn] = __builtin_amdgcn_wmma_f32_16x16x32_bf16(
            false, af[tm], false, bfv[tn], (short)0, acc[tm][tn], false, false);
    __syncthreads();
  }

#pragma unroll
  for (int tm = 0; tm < 2; tm++)
#pragma unroll
    for (int tn = 0; tn < 4; tn++) {
      const int col = bn0 + wn * 64 + tn * 16 + lr;
      const float bv = bias[col];
#pragma unroll
      for (int i = 0; i < 8; i++) {
        const int row = bm0 + wm * 32 + tm * 16 + hl * 8 + i;
        float v = acc[tm][tn][i] + bv;
        if (resid) v += resid[(size_t)row * N + col];
        if (outf) outf[(size_t)row * N + col] = v;
        if (outb) outb[(size_t)row * N + col] = (__bf16)v;
      }
    }
}

// ---------------------------------------------------------------------------
// Scores: S[b,h,q,k] = scale * Q_h K_h^T, Q/K already bf16.
// One block per (b,h). K staged to LDS (pure copy); Q fragments are direct
// 16B global loads matching the WMMA A layout.
// ---------------------------------------------------------------------------
__global__ __launch_bounds__(256)
void attn_scores_kernel(const __bf16* __restrict__ Q, const __bf16* __restrict__ Kmat,
                        float* __restrict__ S) {
  constexpr int ST = DK_ + 8;       // 72
  __shared__ __bf16 Ks[N_ * ST];
  const int tid = threadIdx.x, lane = tid & 31, wid = tid >> 5;
  const int lr = lane & 15, hl = lane >> 4;
  const int bh = blockIdx.x, b = bh >> 4, h = bh & 15;
  const __bf16* Qb = Q + (size_t)b * N_ * D_ + h * DK_;
  const __bf16* Kb = Kmat + (size_t)b * N_ * D_ + h * DK_;

  {
    const int c8 = (tid & 7) * 8, r0 = tid >> 3;  // r0 in 0..31
#pragma unroll
    for (int p = 0; p < 8; p++) {
      const int key = r0 + p * 32;
      *(v8bf*)&Ks[key * ST + c8] = *(const v8bf*)(Kb + (size_t)key * D_ + c8);
    }
  }
  __syncthreads();

  v16bf af[2][2];
#pragma unroll
  for (int ks = 0; ks < 2; ks++)
#pragma unroll
    for (int t = 0; t < 2; t++) {
      const int row = wid * 32 + t * 16 + lr;
      const __bf16* pl = Qb + (size_t)row * D_ + ks * 32 + hl * 8;
      const v8bf lo = *(const v8bf*)(pl);
      const v8bf hb = *(const v8bf*)(pl + 16);
      af[ks][t] = cat8(lo, hb);
    }

  const float scale = 0.125f;  // 1/sqrt(64)
  float* Sb = S + (size_t)bh * N_ * N_;
  for (int nb = 0; nb < 4; nb++) {
    v8f acc[2][4];
#pragma unroll
    for (int i = 0; i < 2; i++)
#pragma unroll
      for (int j = 0; j < 4; j++) acc[i][j] = zero_v8f();
#pragma unroll
    for (int ks = 0; ks < 2; ks++) {
      v16bf bfv[4];
#pragma unroll
      for (int t = 0; t < 4; t++) {
        const int key = nb * 64 + t * 16 + lr;
        v8bf lo = *(const v8bf*)&Ks[key * ST + ks * 32 + hl * 8];
        v8bf hb = *(const v8bf*)&Ks[key * ST + ks * 32 + 16 + hl * 8];
        bfv[t] = cat8(lo, hb);
      }
#pragma unroll
      for (int tm = 0; tm < 2; tm++)
#pragma unroll
        for (int tn = 0; tn < 4; tn++)
          acc[tm][tn] = __builtin_amdgcn_wmma_f32_16x16x32_bf16(
              false, af[ks][tm], false, bfv[tn], (short)0, acc[tm][tn], false, false);
    }
#pragma unroll
    for (int tm = 0; tm < 2; tm++)
#pragma unroll
      for (int tn = 0; tn < 4; tn++)
#pragma unroll
        for (int i = 0; i < 8; i++) {
          const int qr = wid * 32 + tm * 16 + hl * 8 + i;
          const int kc = nb * 64 + tn * 16 + lr;
          Sb[(size_t)qr * N_ + kc] = acc[tm][tn][i] * scale;
        }
  }
}

// ---------------------------------------------------------------------------
// Row-wise masked softmax * attn_bias, in place over the fp32 p buffer.
// One wave32 per row; 8 rows per block.
// ---------------------------------------------------------------------------
__global__ __launch_bounds__(256)
void softmax_bias_kernel(float* __restrict__ P, const unsigned char* __restrict__ cmask,
                         const float* __restrict__ attn_bias) {
  const int lane = threadIdx.x & 31, wid = threadIdx.x >> 5;
  const size_t row = (size_t)blockIdx.x * 8 + wid;   // over B*H*N
  const int q = (int)(row & (N_ - 1));
  const size_t bh = row >> 8;
  const int h = (int)(bh & (H_ - 1));
  const int b = (int)(bh >> 4);
  float* pr = P + row * N_;
  const float* br = attn_bias + ((size_t)h * N_ + q) * N_;
  const unsigned char* mr = cmask + (size_t)b * N_;
  const float NEG = -__builtin_inff();

  float v[8];
  float mx = NEG;
#pragma unroll
  for (int j = 0; j < 8; j++) {
    const int k = lane + j * 32;
    const float s = mr[k] ? pr[k] : NEG;
    v[j] = s;
    mx = fmaxf(mx, s);
  }
#pragma unroll
  for (int off = 16; off > 0; off >>= 1) mx = fmaxf(mx, __shfl_xor(mx, off, 32));
  float sum = 0.f;
#pragma unroll
  for (int j = 0; j < 8; j++) {
    const float e = (v[j] == NEG) ? 0.f : __expf(v[j] - mx);
    v[j] = e;
    sum += e;
  }
#pragma unroll
  for (int off = 16; off > 0; off >>= 1) sum += __shfl_xor(sum, off, 32);
  const float inv = sum > 0.f ? 1.f / sum : 0.f;
#pragma unroll
  for (int j = 0; j < 8; j++) {
    const int k = lane + j * 32;
    pr[k] = v[j] * inv * br[k];
  }
}

// ---------------------------------------------------------------------------
// O_h = P * V_h, O written as bf16 (feeds Wo GEMM). One block per (b,h).
// V (bf16) staged transposed [dk][key]; p streamed in 32-key LDS tiles.
// ---------------------------------------------------------------------------
__global__ __launch_bounds__(256)
void attn_pv_kernel(const float* __restrict__ P, const __bf16* __restrict__ V,
                    __bf16* __restrict__ O) {
  constexpr int VST = N_ + 8;   // 264
  constexpr int PST = 40;
  __shared__ __bf16 Vt[DK_ * VST];
  __shared__ __bf16 Ps[N_ * PST];
  const int tid = threadIdx.x, lane = tid & 31, wid = tid >> 5;
  const int lr = lane & 15, hl = lane >> 4;
  const int bh = blockIdx.x, b = bh >> 4, h = bh & 15;
  const __bf16* Vb = V + (size_t)b * N_ * D_ + h * DK_;
  const float* Pb = P + (size_t)bh * N_ * N_;

  {
    const int d8 = (tid & 7) * 8, k0 = tid >> 3;  // k0 in 0..31
#pragma unroll
    for (int p = 0; p < 8; p++) {
      const int key = k0 + p * 32;
      const v8bf vv = *(const v8bf*)(Vb + (size_t)key * D_ + d8);
#pragma unroll
      for (int j = 0; j < 8; j++) Vt[(d8 + j) * VST + key] = vv[j];
    }
  }

  v8f acc[2][4];
#pragma unroll
  for (int i = 0; i < 2; i++)
#pragma unroll
    for (int j = 0; j < 4; j++) acc[i][j] = zero_v8f();

  for (int ks = 0; ks < N_; ks += 32) {
    __syncthreads();
    {
      const int c = (tid & 7) * 4, r0 = tid >> 3;
#pragma unroll
      for (int p = 0; p < 8; p++) {
        const int r = r0 + p * 32;
        const float4 pv = *(const float4*)(Pb + (size_t)r * N_ + ks + c);
        __bf16* d = &Ps[r * PST + c];
        d[0] = (__bf16)pv.x; d[1] = (__bf16)pv.y; d[2] = (__bf16)pv.z; d[3] = (__bf16)pv.w;
      }
    }
    __syncthreads();
    v16bf af[2], bfv[4];
#pragma unroll
    for (int t = 0; t < 2; t++) {
      const int row = wid * 32 + t * 16 + lr;
      v8bf lo = *(const v8bf*)&Ps[row * PST + hl * 8];
      v8bf hb = *(const v8bf*)&Ps[row * PST + 16 + hl * 8];
      af[t] = cat8(lo, hb);
    }
#pragma unroll
    for (int t = 0; t < 4; t++) {
      const int dc = t * 16 + lr;
      v8bf lo = *(const v8bf*)&Vt[dc * VST + ks + hl * 8];
      v8bf hb = *(const v8bf*)&Vt[dc * VST + ks + 16 + hl * 8];
      bfv[t] = cat8(lo, hb);
    }
#pragma unroll
    for (int tm = 0; tm < 2; tm++)
#pragma unroll
      for (int tn = 0; tn < 4; tn++)
        acc[tm][tn] = __builtin_amdgcn_wmma_f32_16x16x32_bf16(
            false, af[tm], false, bfv[tn], (short)0, acc[tm][tn], false, false);
  }

#pragma unroll
  for (int tm = 0; tm < 2; tm++)
#pragma unroll
    for (int tn = 0; tn < 4; tn++)
#pragma unroll
      for (int i = 0; i < 8; i++) {
        const int qr = wid * 32 + tm * 16 + hl * 8 + i;
        const int dc = tn * 16 + lr;
        O[((size_t)b * N_ + qr) * D_ + h * DK_ + dc] = (__bf16)acc[tm][tn][i];
      }
}

// ---------------------------------------------------------------------------
// BatchNorm stats: per-feature mean / biased variance over B*N rows.
// ---------------------------------------------------------------------------
__global__ __launch_bounds__(256)
void bn_stats_kernel(const float* __restrict__ X, float* __restrict__ mu,
                     float* __restrict__ var) {
  __shared__ float sh[2][4][64];
  const int c64 = threadIdx.x & 63, rp = threadIdx.x >> 6;
  const int c = blockIdx.x * 64 + c64;
  float s = 0.f, s2 = 0.f;
  for (int r = rp; r < B_ * N_; r += 4) {
    const float v = X[(size_t)r * D_ + c];
    s += v;
    s2 += v * v;
  }
  sh[0][rp][c64] = s;
  sh[1][rp][c64] = s2;
  __syncthreads();
  if (rp == 0) {
    const float ts = sh[0][0][c64] + sh[0][1][c64] + sh[0][2][c64] + sh[0][3][c64];
    const float t2 = sh[1][0][c64] + sh[1][1][c64] + sh[1][2][c64] + sh[1][3][c64];
    const float m = ts / (float)(B_ * N_);
    mu[c] = m;
    var[c] = t2 / (float)(B_ * N_) - m * m;
  }
}

// ---------------------------------------------------------------------------
// Fused normalize + mean-pool (mean over N is linear -> pool then affine).
// ---------------------------------------------------------------------------
__global__ __launch_bounds__(256)
void bn_pool_kernel(const float* __restrict__ X, const float* __restrict__ mu,
                    const float* __restrict__ var, const float* __restrict__ gamma,
                    const float* __restrict__ beta, float* __restrict__ pooled) {
  __shared__ float sh[4][64];
  const int c64 = threadIdx.x & 63, np = threadIdx.x >> 6;
  const int b = blockIdx.x >> 4;
  const int c = (blockIdx.x & 15) * 64 + c64;
  float s = 0.f;
  for (int n = np; n < N_; n += 4) s += X[((size_t)b * N_ + n) * D_ + c];
  sh[np][c64] = s;
  __syncthreads();
  if (np == 0) {
    const float ts = sh[0][c64] + sh[1][c64] + sh[2][c64] + sh[3][c64];
    const float m = ts / (float)N_;
    pooled[(size_t)b * D_ + c] =
        (m - mu[c]) * rsqrtf(var[c] + EPS_) * gamma[c] + beta[c];
  }
}

// ---------------------------------------------------------------------------
extern "C" void kernel_launch(void* const* d_in, const int* in_sizes, int n_in,
                              void* d_out, int out_size, void* d_ws, size_t ws_size,
                              hipStream_t stream) {
  (void)in_sizes; (void)n_in; (void)out_size; (void)ws_size;
  const float* x = (const float*)d_in[0];
  const unsigned char* cmask = (const unsigned char*)d_in[1];  // bool (1 byte)
  const float* Wq = (const float*)d_in[2];
  const float* bq = (const float*)d_in[3];
  const float* Wk = (const float*)d_in[4];
  const float* bk = (const float*)d_in[5];
  const float* Wv = (const float*)d_in[6];
  const float* bv = (const float*)d_in[7];
  const float* Wo = (const float*)d_in[8];
  const float* bo = (const float*)d_in[9];
  const float* attn_bias = (const float*)d_in[10];
  const float* bn_gamma = (const float*)d_in[11];
  const float* bn_beta = (const float*)d_in[12];

  float* pooled = (float*)d_out;
  float* p0 = pooled + (size_t)B_ * D_;
  float* p1 = p0 + (size_t)B_ * H_ * N_ * N_;

  // Workspace layout (~304 MB):
  //   8 x 2MB  bf16 transposed weights
  //   1 x 32MB bf16 activations (GEMM A operand, refreshed per layer)
  //   4 x 32MB bf16 Q/K/V/O
  //   2 x 64MB fp32 residual-stream outputs L0/L1
  //   BN stats
  char* ws = (char*)d_ws;
  const size_t WTE = (size_t)D_ * D_;                 // elements per weight matrix
  const size_t ACT = (size_t)B_ * N_ * D_;            // activation elements
  __bf16* wt = (__bf16*)ws;                           // 8 matrices
  __bf16* Ab = (__bf16*)(ws + 8 * WTE * sizeof(__bf16));
  __bf16* Qb = Ab + ACT;
  __bf16* Kb = Qb + ACT;
  __bf16* Vb = Kb + ACT;
  __bf16* Ob = Vb + ACT;
  float*  L0 = (float*)(Ob + ACT);
  float*  L1 = L0 + ACT;
  float*  mu = L1 + ACT;
  float*  var = mu + D_;

  const int M = B_ * N_;
  const dim3 gemm_grid(D_ / 128, M / 128);  // (8, 128)
  const dim3 tr_grid(16, 16);

  // One-time weight transpose+convert: order [l][q,k,v,o].
  const float* Wsrc[8] = {Wq, Wk, Wv, Wo,
                          Wq + WTE, Wk + WTE, Wv + WTE, Wo + WTE};
  for (int i = 0; i < 8; i++)
    wt_transpose_kernel<<<tr_grid, 256, 0, stream>>>(Wsrc[i], wt + (size_t)i * WTE);

  // x -> bf16 activations for layer 0.
  cvt_bf16_kernel<<<(int)(ACT / 2048), 256, 0, stream>>>(x, Ab);

  for (int l = 0; l < 2; l++) {
    const float* curf = (l == 0) ? x : L0;   // fp32 residual stream
    float* nxt = (l == 0) ? L0 : L1;
    float* P = (l == 0) ? p0 : p1;
    const __bf16* wqt = wt + (size_t)(4 * l + 0) * WTE;
    const __bf16* wkt = wt + (size_t)(4 * l + 1) * WTE;
    const __bf16* wvt = wt + (size_t)(4 * l + 2) * WTE;
    const __bf16* wot = wt + (size_t)(4 * l + 3) * WTE;
    const float* bql = bq + (size_t)l * D_;
    const float* bkl = bk + (size_t)l * D_;
    const float* bvl = bv + (size_t)l * D_;
    const float* bol = bo + (size_t)l * D_;

    // Q/K/V projections: bf16-only outputs.
    gemm_bias_kernel<<<gemm_grid, 256, 0, stream>>>(Ab, wqt, bql, nullptr, nullptr, Qb, M, D_, D_);
    gemm_bias_kernel<<<gemm_grid, 256, 0, stream>>>(Ab, wkt, bkl, nullptr, nullptr, Kb, M, D_, D_);
    gemm_bias_kernel<<<gemm_grid, 256, 0, stream>>>(Ab, wvt, bvl, nullptr, nullptr, Vb, M, D_, D_);
    attn_scores_kernel<<<B_ * H_, 256, 0, stream>>>(Qb, Kb, P);
    softmax_bias_kernel<<<(B_ * H_ * N_) / 8, 256, 0, stream>>>(P, cmask, attn_bias);
    attn_pv_kernel<<<B_ * H_, 256, 0, stream>>>(P, Vb, Ob);
    // Output projection + residual: fp32 stream AND bf16 copy (= next layer's A).
    gemm_bias_kernel<<<gemm_grid, 256, 0, stream>>>(Ob, wot, bol, curf, nxt, Ab, M, D_, D_);
  }

  bn_stats_kernel<<<D_ / 64, 256, 0, stream>>>(L1, mu, var);
  bn_pool_kernel<<<B_ * 16, 256, 0, stream>>>(L1, mu, var, bn_gamma, bn_beta, pooled);
}
